// ARRPSRT_5660766896388
// MI455X (gfx1250) — matrix operand order repaired and checked by reading, not compiled, not run on hardware
//
#include <hip/hip_runtime.h>
#include <hip/hip_bf16.h>
#include <math.h>
#include <stdint.h>

typedef _Float16 half_t;
typedef __attribute__((ext_vector_type(16))) _Float16 v16h;
typedef __attribute__((ext_vector_type(8)))  float    v8f;
typedef __attribute__((ext_vector_type(4)))  unsigned int v4u;
typedef __attribute__((ext_vector_type(8)))  int      v8i;
typedef __attribute__((ext_vector_type(4)))  int      v4i;

#define B_  2
#define L_  2048
#define D_  1024
#define H_  16
#define KV_ 4
#define DH_ 64
#define E_  3
#define FF_ 3072
#define MB_ 72

union Frag16 {
    v16h   v;
    half_t h[16];
    uint4  u[2];
};

// ---------------- TDM helper ----------------
// 2D tensor tile load Global -> LDS via Tensor Data Mover.
// D# built per CDNA5 ISA 8.3/8.4: group0 {count=1, lds_addr, global_addr, type=2},
// group1 {data_size=2B, pad_enable, pad codes, tensor dims, tile dims, dim0 stride}.
// OOB rows (>= tensor_d1) read as zero. LDS padding: pad_amt dwords after every
// pad_int-coded dwords (applied to LDS addresses only).
// NOTE: this toolchain exposes the 6-arg builtin:
//   (uint32x4 g0, int32x8 g1, int32x4 g2, int32x4 g3, int32x8 g4, i32 cpol)
__device__ __forceinline__ void tdm_load_2d(
    unsigned lds_addr, const void* gptr,
    unsigned tile_d0, unsigned tile_d1,
    unsigned tensor_d0, unsigned tensor_d1,
    unsigned stride_d0,
    unsigned pad_int_code, unsigned pad_amt_code)
{
    unsigned long long ga = (unsigned long long)(uintptr_t)gptr;
    v4u g0;
    g0[0] = 1u;                                            // count=1 (user D#)
    g0[1] = lds_addr;                                      // lds_addr[31:0]
    g0[2] = (unsigned)ga;                                  // global_addr[31:0]
    g0[3] = (unsigned)((ga >> 32) & 0x1FFFFFFu) | (2u << 30); // ga[56:32] | type=2
    v8i g1;
    g1[0] = (int)((1u << 16)                               // data_size = 2 bytes
                | (1u << 20)                               // pad_enable
                | (pad_int_code << 22)                     // pad_interval
                | (pad_amt_code << 25));                   // pad_amount
    g1[1] = (int)((tensor_d0 & 0xFFFFu) << 16);            // tensor_dim0[15:0]
    g1[2] = (int)(((tensor_d0 >> 16) & 0xFFFFu)            // tensor_dim0[31:16]
                | ((tensor_d1 & 0xFFFFu) << 16));          // tensor_dim1[15:0]
    g1[3] = (int)(((tensor_d1 >> 16) & 0xFFFFu)            // tensor_dim1[31:16]
                | ((tile_d0 & 0xFFFFu) << 16));            // tile_dim0
    g1[4] = (int)(tile_d1 & 0xFFFFu);                      // tile_dim1 (tile_dim2=0)
    g1[5] = (int)stride_d0;                                // tensor_dim0_stride[31:0]
    g1[6] = 0;                                             // stride hi / dim1_stride lo
    g1[7] = 0;
    v4i g2 = {0, 0, 0, 0};
    v4i g3 = {0, 0, 0, 0};
    v8i g4 = {0, 0, 0, 0, 0, 0, 0, 0};
    __builtin_amdgcn_tensor_load_to_lds(g0, g1, g2, g3, g4, 0);
}

// ---------------- elementwise helpers ----------------

__global__ void k_copy_f32(const float* __restrict__ src, float* __restrict__ dst, int n) {
    int i = blockIdx.x * blockDim.x + threadIdx.x;
    int st = gridDim.x * blockDim.x;
    for (; i < n; i += st) dst[i] = src[i];
}

__global__ void k_cast_f16(const float* __restrict__ src, half_t* __restrict__ dst, int n) {
    int i = blockIdx.x * blockDim.x + threadIdx.x;
    int st = gridDim.x * blockDim.x;
    for (; i < n; i += st) dst[i] = (half_t)src[i];
}

__global__ void k_silu_mul(half_t* __restrict__ g, const half_t* __restrict__ u, int n) {
    int i = blockIdx.x * blockDim.x + threadIdx.x;
    int st = gridDim.x * blockDim.x;
    for (; i < n; i += st) {
        float a = (float)g[i], b = (float)u[i];
        g[i] = (half_t)((a / (1.f + __expf(-a))) * b);
    }
}

// transpose-cast: src f32 [K][N] -> dst f16 [N][K]; grid (N/32, K/32), 256 thr
__global__ __launch_bounds__(256) void k_cast_transpose(
    const float* __restrict__ src, half_t* __restrict__ dst, int K, int N)
{
    __shared__ float tile[32][33];
    const int kb = blockIdx.y * 32, nb = blockIdx.x * 32;
    const int tx = threadIdx.x & 31;
    const int wid = threadIdx.x >> 5;
    #pragma unroll
    for (int r = 0; r < 4; ++r) {
        int k = wid * 4 + r;
        tile[k][tx] = src[(size_t)(kb + k) * N + nb + tx];
    }
    __syncthreads();
    #pragma unroll
    for (int r = 0; r < 4; ++r) {
        int n = wid * 4 + r;
        dst[(size_t)(nb + n) * K + kb + tx] = (half_t)tile[tx][n];
    }
}

// RMSNorm: one block (256 thr = 8 waves) per row of D_=1024
__global__ __launch_bounds__(256) void k_rmsnorm_f16(
    const float* __restrict__ x, const float* __restrict__ w,
    half_t* __restrict__ out)
{
    __shared__ float red[8];
    const int row = blockIdx.x;
    const float* xr = x + (size_t)row * D_;
    float ss = 0.f;
    for (int i = threadIdx.x; i < D_; i += 256) { float v = xr[i]; ss += v * v; }
    #pragma unroll
    for (int m = 16; m >= 1; m >>= 1) ss += __shfl_xor(ss, m, 32);
    const int wid = threadIdx.x >> 5;
    if ((threadIdx.x & 31) == 0) red[wid] = ss;
    __syncthreads();
    if (wid == 0) {
        float v = (threadIdx.x < 8) ? red[threadIdx.x] : 0.f;
        #pragma unroll
        for (int m = 4; m >= 1; m >>= 1) v += __shfl_xor(v, m, 32);
        if (threadIdx.x == 0) red[0] = v;
    }
    __syncthreads();
    const float inv = rsqrtf(red[0] * (1.0f / D_) + 1e-5f);
    half_t* orow = out + (size_t)row * D_;
    for (int i = threadIdx.x; i < D_; i += 256) orow[i] = (half_t)(xr[i] * inv * w[i]);
}

// RoPE in-place on [B, nh, L, 64] f16
__global__ void k_rope(half_t* __restrict__ x, const float* __restrict__ cs,
                       const float* __restrict__ sn, int total)
{
    int tid = blockIdx.x * blockDim.x + threadIdx.x;
    if (tid >= total) return;
    int d = tid & 31;
    int l = (tid >> 5) & (L_ - 1);
    int bh = tid >> 16;
    half_t* p = x + ((size_t)bh * L_ + l) * 64;
    float x1 = (float)p[d], x2 = (float)p[d + 32];
    float c = cs[l * 32 + d], s = sn[l * 32 + d];
    p[d]      = (half_t)(x1 * c - x2 * s);
    p[d + 32] = (half_t)(x2 * c + x1 * s);
}

// ---------------- WMMA GEMM (TDM double-buffered) ----------------
// C[M,N] = A[M,K] @ B[K,N] with B stored TRANSPOSED as BwT[N][K] (f16).
// Block: 256 thr = 8 waves; block tile 256(M) x 64(N); wave tile 32 x 64
// (2 A-frags x 4 B-frags = 8 WMMAs per 32-wide k-step).
// B tile [64 n][32 k] is DMA'd into LDS by the Tensor Data Mover (wave 0),
// double-buffered against compute; TDM pad fields produce the 40-half row
// stride (20-bank spread) for conflict-free ds_load_b128 fragment reads.
__global__ __launch_bounds__(256) void k_gemm_f16(
    const half_t* __restrict__ A, const half_t* __restrict__ BwT,
    half_t* __restrict__ outH, float* __restrict__ outF,
    int Mrows, int N, int K,
    int nheads, int Lper,
    const float* __restrict__ gate,
    const float* __restrict__ ew, const float* __restrict__ beta, int eidx)
{
    __shared__ __align__(16) half_t BsT[2][64][40];
    const int lane = threadIdx.x & 31;
    const int wv   = threadIdx.x >> 5;
    const int rowbase = blockIdx.y * 256 + wv * 32;
    const int c0      = blockIdx.x * 64;

    v8f acc[2][4] = {};

    const int nl    = lane & 15;
    const int abase = (lane >> 4) << 3;     // A-frag half-wave K offset (0/8)
    const int kk0   = (lane >> 4) << 4;     // unused; B-frag also uses abase (0/8)
    const int aRow0 = rowbase + nl;
    const int aRow1 = rowbase + 16 + nl;
    const bool aOk0 = aRow0 < Mrows;
    const bool aOk1 = aRow1 < Mrows;

    const half_t* gB = BwT + (size_t)c0 * K;   // tile origin: rows c0..c0+63

    // TDM stage of tile 0 (wave 0 issues for the whole block)
    if (wv == 0) {
        tdm_load_2d((unsigned)(uintptr_t)&BsT[0][0][0], gB,
                    /*tile*/ 32, 64, /*tensor*/ (unsigned)K, 64,
                    (unsigned)K, /*pad: every 16 dwords*/ 3, /*+4 dwords*/ 3);
    }

    int buf = 0;
    for (int k0 = 0; k0 < K; k0 += 32) {
        if (wv == 0) {
            if (k0 + 32 < K) {
                tdm_load_2d((unsigned)(uintptr_t)&BsT[buf ^ 1][0][0], gB + k0 + 32,
                            32, 64, (unsigned)K, 64, (unsigned)K, 3, 3);
                __builtin_amdgcn_s_wait_tensorcnt(1);   // current tile done
            } else {
                __builtin_amdgcn_s_wait_tensorcnt(0);
            }
        }
        __syncthreads();

        Frag16 af0, af1;
        if (aOk0) {
            const half_t* ap = A + (size_t)aRow0 * K + k0 + abase;
            af0.u[0] = *(const uint4*)(ap);
            af0.u[1] = *(const uint4*)(ap + 16);
            if (k0 + 32 < K) __builtin_prefetch(ap + 32, 0, 0);
        } else { af0.u[0] = make_uint4(0,0,0,0); af0.u[1] = make_uint4(0,0,0,0); }
        if (aOk1) {
            const half_t* ap = A + (size_t)aRow1 * K + k0 + abase;
            af1.u[0] = *(const uint4*)(ap);
            af1.u[1] = *(const uint4*)(ap + 16);
        } else { af1.u[0] = make_uint4(0,0,0,0); af1.u[1] = make_uint4(0,0,0,0); }

        #pragma unroll
        for (int j = 0; j < 4; ++j) {
            Frag16 bf;
            const half_t* bp = &BsT[buf][j * 16 + nl][abase];
            bf.u[0] = *(const uint4*)(bp);
            bf.u[1] = *(const uint4*)(bp + 16);
            acc[0][j] = __builtin_amdgcn_wmma_f32_16x16x32_f16(
                false, af0.v, false, bf.v, (short)0, acc[0][j], false, false);
            acc[1][j] = __builtin_amdgcn_wmma_f32_16x16x32_f16(
                false, af1.v, false, bf.v, (short)0, acc[1][j], false, false);
        }
        __syncthreads();
        buf ^= 1;
    }

    // epilogue: C element (v, lane) -> row = t*16 + v + 8*(lane>>4), col = lane&15
    float gscale = 1.f;
    if (outF && gate) gscale = 1.f / (1.f + __expf(-gate[0]));
    const int rlo = rowbase + 8 * (lane >> 4);
    #pragma unroll
    for (int t = 0; t < 2; ++t) {
        #pragma unroll
        for (int j = 0; j < 4; ++j) {
            #pragma unroll
            for (int v = 0; v < 8; ++v) {
                int row = rlo + t * 16 + v;
                if (row >= Mrows) continue;
                int col = c0 + j * 16 + nl;
                float val = acc[t][j][v];
                if (outH) {
                    size_t idx;
                    if (nheads > 0) {
                        int b = row / Lper, l = row - b * Lper;
                        int h = col >> 6, d = col & 63;
                        idx = (((size_t)(b * nheads + h)) * Lper + l) * 64 + d;
                    } else {
                        idx = (size_t)row * N + col;
                    }
                    outH[idx] = (half_t)val;
                } else {
                    float s = gscale;
                    if (ew) { int b = row / Lper; s *= ew[b * E_ + eidx] * beta[eidx]; }
                    outF[(size_t)row * N + col] += s * val;
                }
            }
        }
    }
}

// ---------------- Flash attention (WMMA + per-wave TDM) ----------------
// q: [B, H_, Lq, 64], k/v: [B, HK, Lk, 64] f16, out: [B, Lq, H_*64] f16.
// Block 128 thr = 4 waves; each wave owns one 16-query tile and DMAs its own
// V tile via TDM (OOB rows beyond Lk auto-zero); DMA overlaps QK^T + softmax.
__global__ __launch_bounds__(128) void k_flash(
    const half_t* __restrict__ qh, const half_t* __restrict__ kh,
    const half_t* __restrict__ vh, half_t* __restrict__ oh,
    int Lq, int Lk, int HK, int causal)
{
    __shared__ __align__(16) half_t Ps[4][16][32];
    __shared__ __align__(16) half_t Vs[4][32][72];   // 72-half stride via TDM padding
    const int lane = threadIdx.x & 31;
    const int wv   = threadIdx.x >> 5;
    const int h    = blockIdx.y;
    const int b    = blockIdx.z;
    const int qtile = blockIdx.x * 4 + wv;
    const int qbase = qtile * 16;
    const int hk = h / (H_ / HK);

    const half_t* qhp = qh + (((size_t)(b * H_ + h)) * Lq) * 64;
    const half_t* khp = kh + (((size_t)(b * HK + hk)) * Lk) * 64;
    const half_t* vhp = vh + (((size_t)(b * HK + hk)) * Lk) * 64;

    const int g  = lane >> 4;
    const int nl = lane & 15;

    Frag16 qf0, qf1;
    {
        const half_t* qp = qhp + (size_t)(qbase + nl) * 64 + (g << 3);
        qf0.u[0] = *(const uint4*)(qp);
        qf0.u[1] = *(const uint4*)(qp + 16);
        qf1.u[0] = *(const uint4*)(qp + 32);
        qf1.u[1] = *(const uint4*)(qp + 48);
    }

    float runm[8], lsum[8];
    #pragma unroll
    for (int v = 0; v < 8; ++v) { runm[v] = -1e30f; lsum[v] = 0.f; }
    v8f O[4] = {};

    int kend = causal ? (qbase + 16) : Lk;
    if (kend > Lk) kend = Lk;

    const unsigned vs_lds = (unsigned)(uintptr_t)&Vs[wv][0][0];

    for (int kt = 0; kt < kend; kt += 32) {
        // per-wave TDM: V tile [32 keys][64 dh] -> LDS; rows >= Lk-kt read zero.
        // tile row = 128B = 32 dwords -> pad_interval code 4, pad 4 dwords -> 72-half stride
        tdm_load_2d(vs_lds, vhp + (size_t)kt * 64,
                    /*tile*/ 64, 32, /*tensor*/ 64, (unsigned)(Lk - kt),
                    /*stride*/ 64, /*pad codes*/ 4, 3);

        // S = Q K^T while the V DMA flies
        v8f s[2];
        #pragma unroll
        for (int n0i = 0; n0i < 2; ++n0i) {
            int key = kt + n0i * 16 + nl;
            Frag16 b0, b1;
            if (key < Lk) {
                const half_t* kp = khp + (size_t)key * 64 + (g << 3);
                b0.u[0] = *(const uint4*)(kp);
                b0.u[1] = *(const uint4*)(kp + 16);
                b1.u[0] = *(const uint4*)(kp + 32);
                b1.u[1] = *(const uint4*)(kp + 48);
            } else {
                b0.u[0] = b0.u[1] = b1.u[0] = b1.u[1] = make_uint4(0, 0, 0, 0);
            }
            v8f a = {};
            a = __builtin_amdgcn_wmma_f32_16x16x32_f16(false, qf0.v, false, b0.v, (short)0, a, false, false);
            a = __builtin_amdgcn_wmma_f32_16x16x32_f16(false, qf1.v, false, b1.v, (short)0, a, false, false);
            s[n0i] = a;
        }

        // online softmax; row of element (v,lane) = qbase + v + 8*g
        #pragma unroll
        for (int v = 0; v < 8; ++v) {
            int q = qbase + v + 8 * g;
            float s0 = s[0][v] * 0.125f;
            float s1 = s[1][v] * 0.125f;
            int k0i = kt + nl, k1i = kt + 16 + nl;
            if (k0i >= Lk || (causal && k0i > q)) s0 = -1e30f;
            if (k1i >= Lk || (causal && k1i > q)) s1 = -1e30f;
            float lm = fmaxf(s0, s1);
            #pragma unroll
            for (int m = 8; m >= 1; m >>= 1) lm = fmaxf(lm, __shfl_xor(lm, m, 32));
            float nm   = fmaxf(runm[v], lm);
            float corr = __expf(runm[v] - nm);
            float p0   = __expf(s0 - nm);
            float p1   = __expf(s1 - nm);
            float rs   = p0 + p1;
            #pragma unroll
            for (int m = 8; m >= 1; m >>= 1) rs += __shfl_xor(rs, m, 32);
            lsum[v] = lsum[v] * corr + rs;
            runm[v] = nm;
            #pragma unroll
            for (int j = 0; j < 4; ++j) O[j][v] *= corr;
            Ps[wv][v + 8 * g][nl]      = (half_t)p0;
            Ps[wv][v + 8 * g][16 + nl] = (half_t)p1;
        }
        asm volatile("s_wait_dscnt 0x0" ::: "memory");

        // P (C-layout in LDS) -> A-fragment
        Frag16 pf;
        {
            const half_t* pp = &Ps[wv][nl][g << 3];
            pf.u[0] = *(const uint4*)(pp);
            pf.u[1] = *(const uint4*)(pp + 16);
        }

        // V DMA must have landed before fragment gathers
        __builtin_amdgcn_s_wait_tensorcnt(0);
        asm volatile("" ::: "memory");

        #pragma unroll
        for (int j = 0; j < 4; ++j) {
            Frag16 vf;
            #pragma unroll
            for (int i = 0; i < 16; ++i)
                vf.h[i] = Vs[wv][g * 8 + (i & 7) + (i >> 3) * 16][j * 16 + nl];
            O[j] = __builtin_amdgcn_wmma_f32_16x16x32_f16(
                false, pf.v, false, vf.v, (short)0, O[j], false, false);
        }
    }

    #pragma unroll
    for (int v = 0; v < 8; ++v) {
        float inv = 1.f / lsum[v];
        int q = qbase + v + 8 * g;
        #pragma unroll
        for (int j = 0; j < 4; ++j) {
            int d = j * 16 + nl;
            oh[((size_t)(b * Lq + q)) * (H_ * 64) + h * 64 + d] = (half_t)(O[j][v] * inv);
        }
    }
}

// ---------------- host orchestration ----------------

extern "C" void kernel_launch(void* const* d_in, const int* in_sizes, int n_in,
                              void* d_out, int out_size, void* d_ws, size_t ws_size,
                              hipStream_t stream)
{
    (void)in_sizes; (void)n_in; (void)out_size; (void)ws_size;
    const float* r_in  = (const float*)d_in[0];
    const float* m0    = (const float*)d_in[1];
    const float* bank  = (const float*)d_in[2];
    const float* rcos  = (const float*)d_in[3];
    const float* rsin  = (const float*)d_in[4];
    const float* ew    = (const float*)d_in[5];
    const float* beta  = (const float*)d_in[6];
    const float* rn_w  = (const float*)d_in[7];
    const float* rq    = (const float*)d_in[8];
    const float* rk    = (const float*)d_in[9];
    const float* rv    = (const float*)d_in[10];
    const float* ro    = (const float*)d_in[11];
    const float* sn_w  = (const float*)d_in[12];
    const float* sq    = (const float*)d_in[13];
    const float* sk    = (const float*)d_in[14];
    const float* sv    = (const float*)d_in[15];
    const float* so    = (const float*)d_in[16];
    const float* mn_w  = (const float*)d_in[17];
    const float* mq    = (const float*)d_in[18];
    const float* mk    = (const float*)d_in[19];
    const float* mv    = (const float*)d_in[20];
    const float* mo    = (const float*)d_in[21];
    const float* mgate = (const float*)d_in[22];
    const float* n2_w  = (const float*)d_in[23];
    const float* wg    = (const float*)d_in[24];
    const float* wu    = (const float*)d_in[25];
    const float* wd    = (const float*)d_in[26];

    float* R = (float*)d_out;

    char* wsp = (char*)d_ws;
    auto alloc = [&](size_t n) -> half_t* {
        half_t* p = (half_t*)wsp;
        wsp += (n * sizeof(half_t) + 255) & ~(size_t)255;
        return p;
    };

    const size_t BLD = (size_t)B_ * L_ * D_;
    half_t* xn    = alloc(BLD);
    half_t* m0h   = alloc(BLD);
    half_t* bankh = alloc((size_t)B_ * MB_ * D_);
    half_t* w_rq  = alloc(D_ * H_ * DH_);      // all weight buffers hold W^T [N][K]
    half_t* w_rk  = alloc(D_ * H_ * DH_);
    half_t* w_rv  = alloc(D_ * H_ * DH_);
    half_t* w_ro  = alloc(H_ * DH_ * D_);
    half_t* w_sq  = alloc(D_ * H_ * DH_);
    half_t* w_sk  = alloc(D_ * KV_ * DH_);
    half_t* w_sv  = alloc(D_ * KV_ * DH_);
    half_t* w_so  = alloc(H_ * DH_ * D_);
    half_t* w_mq  = alloc(D_ * H_ * DH_);
    half_t* w_mk  = alloc(D_ * H_ * DH_);
    half_t* w_mv  = alloc(D_ * H_ * DH_);
    half_t* w_mo  = alloc(H_ * DH_ * D_);
    half_t* w_wg  = alloc((size_t)E_ * D_ * FF_);
    half_t* w_wu  = alloc((size_t)E_ * D_ * FF_);
    half_t* w_wd  = alloc((size_t)E_ * FF_ * D_);
    half_t* qh    = alloc((size_t)B_ * H_ * L_ * DH_);
    half_t* khb   = alloc((size_t)B_ * H_ * L_ * DH_);
    half_t* vhb   = alloc((size_t)B_ * H_ * L_ * DH_);
    half_t* oh    = alloc((size_t)B_ * L_ * H_ * DH_);
    half_t* gh    = alloc((size_t)B_ * L_ * FF_);
    half_t* uh    = alloc((size_t)B_ * L_ * FF_);

    auto cast = [&](const float* s, half_t* d, size_t n) {
        int blocks = (int)((n + 1023) / 1024); if (blocks > 4096) blocks = 4096;
        k_cast_f16<<<blocks, 256, 0, stream>>>(s, d, (int)n);
    };
    auto castT = [&](const float* s, half_t* d, int K, int N) {
        dim3 grid(N / 32, K / 32);
        k_cast_transpose<<<grid, 256, 0, stream>>>(s, d, K, N);
    };
    auto gemm = [&](const half_t* A, const half_t* BwT, half_t* oH, float* oF,
                    int M, int N, int K, int nh, int Lper,
                    const float* gate, const float* ewp, const float* betap, int eidx) {
        dim3 grid(N / 64, (M + 255) / 256);
        k_gemm_f16<<<grid, 256, 0, stream>>>(A, BwT, oH, oF, M, N, K, nh, Lper,
                                             gate, ewp, betap, eidx);
    };
    auto flash = [&](const half_t* q, const half_t* k, const half_t* v, half_t* o,
                     int Lk, int HK, int causal) {
        dim3 grid(L_ / 64, H_, B_);
        k_flash<<<grid, 128, 0, stream>>>(q, k, v, o, L_, Lk, HK, causal);
    };
    auto rmsnorm = [&](const float* x, const float* w, half_t* o) {
        k_rmsnorm_f16<<<B_ * L_, 256, 0, stream>>>(x, w, o);
    };

    // residual init + casts (weights transpose-cast to [N][K] f16)
    k_copy_f32<<<4096, 256, 0, stream>>>(r_in, R, (int)BLD);
    cast(m0, m0h, BLD);
    cast(bank, bankh, (size_t)B_ * MB_ * D_);
    castT(rq, w_rq, D_, H_ * DH_);  castT(rk, w_rk, D_, H_ * DH_);
    castT(rv, w_rv, D_, H_ * DH_);  castT(ro, w_ro, H_ * DH_, D_);
    castT(sq, w_sq, D_, H_ * DH_);  castT(sk, w_sk, D_, KV_ * DH_);
    castT(sv, w_sv, D_, KV_ * DH_); castT(so, w_so, H_ * DH_, D_);
    castT(mq, w_mq, D_, H_ * DH_);  castT(mk, w_mk, D_, H_ * DH_);
    castT(mv, w_mv, D_, H_ * DH_);  castT(mo, w_mo, H_ * DH_, D_);
    for (int e = 0; e < E_; ++e) {
        castT(wg + (size_t)e * D_ * FF_, w_wg + (size_t)e * D_ * FF_, D_, FF_);
        castT(wu + (size_t)e * D_ * FF_, w_wu + (size_t)e * D_ * FF_, D_, FF_);
        castT(wd + (size_t)e * FF_ * D_, w_wd + (size_t)e * FF_ * D_, FF_, D_);
    }

    const int BL = B_ * L_;

    // ---- stage 1: bank cross-attention ----
    rmsnorm(R, rn_w, xn);
    gemm(xn,    w_rq, qh,  nullptr, BL,       1024, 1024, H_, L_,  nullptr, nullptr, nullptr, 0);
    gemm(bankh, w_rk, khb, nullptr, B_ * MB_, 1024, 1024, H_, MB_, nullptr, nullptr, nullptr, 0);
    gemm(bankh, w_rv, vhb, nullptr, B_ * MB_, 1024, 1024, H_, MB_, nullptr, nullptr, nullptr, 0);
    flash(qh, khb, vhb, oh, MB_, H_, 0);
    gemm(oh, w_ro, nullptr, R, BL, 1024, 1024, 0, L_, nullptr, nullptr, nullptr, 0);

    // ---- stage 2: GQA causal self-attention with RoPE ----
    rmsnorm(R, sn_w, xn);
    gemm(xn, w_sq, qh,  nullptr, BL, 1024, 1024, H_,  L_, nullptr, nullptr, nullptr, 0);
    gemm(xn, w_sk, khb, nullptr, BL, 256,  1024, KV_, L_, nullptr, nullptr, nullptr, 0);
    gemm(xn, w_sv, vhb, nullptr, BL, 256,  1024, KV_, L_, nullptr, nullptr, nullptr, 0);
    {
        int tq = B_ * H_ * L_ * 32, tk = B_ * KV_ * L_ * 32;
        k_rope<<<(tq + 255) / 256, 256, 0, stream>>>(qh, rcos, rsin, tq);
        k_rope<<<(tk + 255) / 256, 256, 0, stream>>>(khb, rcos, rsin, tk);
    }
    flash(qh, khb, vhb, oh, L_, KV_, 1);
    gemm(oh, w_so, nullptr, R, BL, 1024, 1024, 0, L_, nullptr, nullptr, nullptr, 0);

    // ---- stage 3: gated memory cross-attention ----
    rmsnorm(R, mn_w, xn);
    gemm(xn,  w_mq, qh,  nullptr, BL, 1024, 1024, H_, L_, nullptr, nullptr, nullptr, 0);
    gemm(m0h, w_mk, khb, nullptr, BL, 1024, 1024, H_, L_, nullptr, nullptr, nullptr, 0);
    gemm(m0h, w_mv, vhb, nullptr, BL, 1024, 1024, H_, L_, nullptr, nullptr, nullptr, 0);
    flash(qh, khb, vhb, oh, L_, H_, 0);
    gemm(oh, w_mo, nullptr, R, BL, 1024, 1024, 0, L_, mgate, nullptr, nullptr, 0);

    // ---- stage 4: 3-expert gated FFN ----
    rmsnorm(R, n2_w, xn);
    for (int e = 0; e < E_; ++e) {
        const half_t* wge = w_wg + (size_t)e * D_ * FF_;
        const half_t* wue = w_wu + (size_t)e * D_ * FF_;
        const half_t* wde = w_wd + (size_t)e * FF_ * D_;
        gemm(xn, wge, gh, nullptr, BL, FF_, 1024, 0, L_, nullptr, nullptr, nullptr, 0);
        gemm(xn, wue, uh, nullptr, BL, FF_, 1024, 0, L_, nullptr, nullptr, nullptr, 0);
        k_silu_mul<<<4096, 256, 0, stream>>>(gh, uh, BL * FF_);
        gemm(gh, wde, nullptr, R, BL, 1024, FF_, 0, L_, nullptr, ew, beta, e);
    }
}
